// Fusion_17523466568155
// MI455X (gfx1250) — matrix operand order repaired and checked by reading, not compiled, not run on hardware
//
#include <hip/hip_runtime.h>

// ---------------------------------------------------------------------------
// MI455X / gfx1250: wave32, WMMA 16x16x32 bf16 (fp32 accumulate).
//
// Pipeline:
//   wt_kernel      : fp32 W (KxN) -> bf16 W^T (NxK)  (one-time, 2.3 MB, L2-hot)
//   mlp2_kernel<..>: fused  relu(X@W1+b1)@W2+b2  per channel, WMMA bf16
//       compress:  X=map_raw (C,512,128) -> map_tmp_t (C,256,512) bf16, transposed out
//       denoise :  X=map_tmp_t (C,256,512) -> map_st (C,256,512) fp32
//   fuse_kernel    : gather K=16 rows, L1 scores, softmax, weighted sum, gate blend
// ---------------------------------------------------------------------------

typedef __attribute__((ext_vector_type(16))) __bf16 v16bf;
typedef __attribute__((ext_vector_type(8)))  __bf16 v8bf;
typedef __attribute__((ext_vector_type(8)))  float  v8f;

// 16-bit A-matrix 16x32 fragment: contiguous K order (K = 16*(lane>>4) + s);
// caller passes p already offset by (lane>>4)*16, so the
// fragment is p[0..15].
__device__ __forceinline__ v16bf load_afrag(const __bf16* p) {
  v8bf lo = *(const v8bf*)(p);
  v8bf hi = *(const v8bf*)(p + 8);
  v16bf a;
#pragma unroll
  for (int i = 0; i < 8; ++i) { a[i] = lo[i]; a[i + 8] = hi[i]; }
  return a;
}

__device__ __forceinline__ v8f wmma_bf16(v16bf a, v16bf b, v8f c) {
  // (neg_a, A, neg_b, B, c_mod, C, reuse_a, reuse_b)
  return __builtin_amdgcn_wmma_f32_16x16x32_bf16(false, a, false, b, (short)0, c,
                                                 false, false);
}

// --------------------------- weight transpose ------------------------------
__global__ __launch_bounds__(256) void wt_kernel(const float* __restrict__ W,
                                                 __bf16* __restrict__ Wt,
                                                 int K, int N) {
  int idx = blockIdx.x * 256 + threadIdx.x;
  if (idx >= K * N) return;
  int k = idx / N;
  int n = idx - k * N;
  Wt[(size_t)n * K + k] = (__bf16)W[idx];
}

// --------------------------- fused 2-layer MLP -----------------------------
// X: (batch, M, K1) of TIN.  W1t: (H, K1) bf16.  W2t: (N, H) bf16.
// Y: (batch, M, N) of TOUT, or (batch, N, M) if TROUT.
// Block: 256 threads = 8 waves. M-tile = 32 (two 16-row strips per wave,
// sharing each B fragment -> 2x reuse of weight traffic).
template <typename TIN, typename TOUT, bool TROUT, int K1, int H, int N, int M>
__global__ __launch_bounds__(256) void mlp2_kernel(
    const TIN* __restrict__ X, const __bf16* __restrict__ W1t,
    const float* __restrict__ b1, const __bf16* __restrict__ W2t,
    const float* __restrict__ b2, TOUT* __restrict__ Y) {
  constexpr int MT = 32;
  constexpr int APAD = K1 + 8;  // +16B/row: fragment loads land 4 banks apart
  constexpr int HPAD = H + 8;
  __shared__ __bf16 ldsA[MT * APAD];
  __shared__ __bf16 ldsH[MT * HPAD];

  const int c    = blockIdx.y;
  const int m0   = blockIdx.x * MT;
  const int t    = threadIdx.x;
  const int wv   = t >> 5;
  const int lane = t & 31;
  const int grp  = lane >> 4;  // which K-half this lane covers
  const int ln   = lane & 15;

  // Stage X tile -> bf16 LDS (row-major, padded).
  const TIN* Xc = X + ((size_t)c * M + m0) * K1;
  for (int e = t; e < MT * K1; e += 256) {
    int m = e / K1, k = e - m * K1;  // K1 is a power of two
    ldsA[m * APAD + k] = (__bf16)Xc[e];
  }
  __syncthreads();

  // ---- GEMM1 + bias + ReLU -> ldsH (bf16, A-layout for GEMM2) ----
  const __bf16* arow0 = ldsA + ln * APAD + grp * 16;
  const __bf16* arow1 = arow0 + 16 * APAD;
  for (int ht = wv; ht < H / 16; ht += 8) {
    const int hb = ht * 16;
    // B fragment: row n=hb+ln of W1t (NxK layout) -> 16 contiguous K per lane.
    const __bf16* brow = W1t + (size_t)(hb + ln) * K1 + grp * 16;
    v8f acc0 = {}, acc1 = {};
#pragma unroll 4
    for (int k = 0; k < K1; k += 32) {
      v16bf b = *(const v16bf*)(brow + k);
      acc0 = wmma_bf16(load_afrag(arow0 + k), b, acc0);
      acc1 = wmma_bf16(load_afrag(arow1 + k), b, acc1);
    }
    const float bias = b1[hb + ln];
#pragma unroll
    for (int r = 0; r < 8; ++r) {  // C/D layout: m = r + 8*grp, n = ln
      int m = r + grp * 8;
      float v0 = acc0[r] + bias;
      float v1 = acc1[r] + bias;
      ldsH[m * HPAD + hb + ln]        = (__bf16)(v0 > 0.f ? v0 : 0.f);
      ldsH[(m + 16) * HPAD + hb + ln] = (__bf16)(v1 > 0.f ? v1 : 0.f);
    }
  }
  __syncthreads();

  // ---- GEMM2 + bias -> Y ----
  const __bf16* hrow0 = ldsH + ln * HPAD + grp * 16;
  const __bf16* hrow1 = hrow0 + 16 * HPAD;
  for (int nt = wv; nt < N / 16; nt += 8) {
    const int nb = nt * 16;
    const __bf16* brow = W2t + (size_t)(nb + ln) * H + grp * 16;
    v8f acc0 = {}, acc1 = {};
#pragma unroll 4
    for (int k = 0; k < H; k += 32) {
      v16bf b = *(const v16bf*)(brow + k);
      acc0 = wmma_bf16(load_afrag(hrow0 + k), b, acc0);
      acc1 = wmma_bf16(load_afrag(hrow1 + k), b, acc1);
    }
    const float bias = b2[nb + ln];
    const int n = nb + ln;
#pragma unroll
    for (int r = 0; r < 8; ++r) {
      int mm0 = m0 + r + grp * 8;
      int mm1 = mm0 + 16;
      float v0 = acc0[r] + bias;
      float v1 = acc1[r] + bias;
      if (TROUT) {
        Y[((size_t)c * N + n) * M + mm0] = (TOUT)v0;
        Y[((size_t)c * N + n) * M + mm1] = (TOUT)v1;
      } else {
        Y[((size_t)c * M + mm0) * N + n] = (TOUT)v0;
        Y[((size_t)c * M + mm1) * N + n] = (TOUT)v1;
      }
    }
  }
}

// ------------------ gather + L1 score + softmax + blend --------------------
__global__ __launch_bounds__(256) void fuse_kernel(
    const float* __restrict__ x_loc,   // (B,C,L)
    const float* __restrict__ map_st,  // (C,DCP,L)  L2-resident (33.5 MB)
    const int* __restrict__ indices,   // (B,C,K)
    const float* __restrict__ lamda1,  // (C,1)
    const float* __restrict__ lamda2,  // (1,L)
    float* __restrict__ out) {         // (B,C,L)
  constexpr int L = 512, K = 16, DCP = 256, C = 64;
  constexpr float SCALE = 0.05f;
  const int c = blockIdx.x, b = blockIdx.y;
  const int t = threadIdx.x;

  __shared__ float xs[L];
  __shared__ float rows[K][L];  // the 16 gathered maps, read twice
  __shared__ int   idx[K];
  __shared__ float sc[K];

  const float* xrow = x_loc + ((size_t)b * C + c) * L;
  if (t < K) idx[t] = indices[((size_t)b * C + c) * K + t];
  xs[t]       = xrow[t];
  xs[t + 256] = xrow[t + 256];
  __syncthreads();

  for (int e = t; e < K * L; e += 256) {  // coalesced along L
    int k = e >> 9, l = e & (L - 1);
    rows[k][l] = map_st[((size_t)c * DCP + idx[k]) * L + l];
  }
  __syncthreads();

  // Scores: wave w reduces k = 2w, 2w+1 (wave32 shuffle reduction).
  const int wv = t >> 5, lane = t & 31;
#pragma unroll
  for (int k2 = 0; k2 < 2; ++k2) {
    int k = wv * 2 + k2;
    float s = 0.f;
#pragma unroll
    for (int j = lane; j < L; j += 32) s += fabsf(xs[j] - rows[k][j]);
#pragma unroll
    for (int off = 16; off > 0; off >>= 1) s += __shfl_xor(s, off, 32);
    if (lane == 0) sc[k] = -s * SCALE;
  }
  __syncthreads();

  // Softmax over K=16 (computed redundantly per thread; trivial cost).
  float wk[K];
  float mx = sc[0];
#pragma unroll
  for (int k = 1; k < K; ++k) mx = fmaxf(mx, sc[k]);
  float sum = 0.f;
#pragma unroll
  for (int k = 0; k < K; ++k) { wk[k] = __expf(sc[k] - mx); sum += wk[k]; }
  const float inv = 1.f / sum;
  const float l1c = 1.f / (1.f + __expf(-lamda1[c]));

  for (int l = t; l < L; l += 256) {
    float acc = 0.f;
#pragma unroll
    for (int k = 0; k < K; ++k) acc += wk[k] * rows[k][l];
    acc *= inv;
    float lam = l1c * (1.f / (1.f + __expf(-lamda2[l])));
    out[((size_t)b * C + c) * L + l] = acc * lam + xs[l] * (1.f - lam);
  }
}

// ---------------------------------------------------------------------------
extern "C" void kernel_launch(void* const* d_in, const int* in_sizes, int n_in,
                              void* d_out, int out_size, void* d_ws,
                              size_t ws_size, hipStream_t stream) {
  (void)in_sizes; (void)n_in; (void)out_size; (void)ws_size;
  const float* x_loc   = (const float*)d_in[0];   // (64,64,512)
  const float* map_raw = (const float*)d_in[1];   // (64,512,128)
  const float* cp_w1   = (const float*)d_in[2];   // (128,256)
  const float* cp_b1   = (const float*)d_in[3];
  const float* cp_w2   = (const float*)d_in[4];   // (256,256)
  const float* cp_b2   = (const float*)d_in[5];
  const float* de_w1   = (const float*)d_in[6];   // (512,1024)
  const float* de_b1   = (const float*)d_in[7];
  const float* de_w2   = (const float*)d_in[8];   // (1024,512)
  const float* de_b2   = (const float*)d_in[9];
  const float* lamda1  = (const float*)d_in[10];  // (64,1)
  const float* lamda2  = (const float*)d_in[11];  // (1,512)
  const int*   indices = (const int*)d_in[12];    // (64,64,16)
  float* out = (float*)d_out;

  char* wsp = (char*)d_ws;
  auto alloc = [&](size_t bytes) {
    char* p = wsp;
    wsp += (bytes + 255) & ~(size_t)255;
    return p;
  };
  __bf16* cpw1t = (__bf16*)alloc((size_t)256 * 128 * 2);
  __bf16* cpw2t = (__bf16*)alloc((size_t)256 * 256 * 2);
  __bf16* dew1t = (__bf16*)alloc((size_t)1024 * 512 * 2);
  __bf16* dew2t = (__bf16*)alloc((size_t)512 * 1024 * 2);
  __bf16* map_tmp_t = (__bf16*)alloc((size_t)64 * 256 * 512 * 2);  // (C,256,512)
  float*  map_st    = (float*)alloc((size_t)64 * 256 * 512 * 4);   // (C,256,512)

  // One-time bf16 transposed weights.
  wt_kernel<<<(128 * 256 + 255) / 256, 256, 0, stream>>>(cp_w1, cpw1t, 128, 256);
  wt_kernel<<<(256 * 256 + 255) / 256, 256, 0, stream>>>(cp_w2, cpw2t, 256, 256);
  wt_kernel<<<(512 * 1024 + 255) / 256, 256, 0, stream>>>(de_w1, dew1t, 512, 1024);
  wt_kernel<<<(1024 * 512 + 255) / 256, 256, 0, stream>>>(de_w2, dew2t, 1024, 512);

  // compress: (C,512,128) -> transposed bf16 (C,256,512)
  mlp2_kernel<float, __bf16, true, 128, 256, 256, 512>
      <<<dim3(512 / 32, 64), 256, 0, stream>>>(map_raw, cpw1t, cp_b1, cpw2t,
                                               cp_b2, map_tmp_t);
  // denoise: (C,256,512) -> fp32 (C,256,512)
  mlp2_kernel<__bf16, float, false, 512, 1024, 512, 256>
      <<<dim3(256 / 32, 64), 256, 0, stream>>>(map_tmp_t, dew1t, de_b1, dew2t,
                                               de_b2, map_st);

  fuse_kernel<<<dim3(64, 64), 256, 0, stream>>>(x_loc, map_st, indices, lamda1,
                                                lamda2, out);
}
